// AttentionHead_69630009803054
// MI455X (gfx1250) — matrix operand order repaired and checked by reading, not compiled, not run on hardware
//
#include <hip/hip_runtime.h>
#include <hip/hip_bf16.h>

typedef __attribute__((ext_vector_type(16))) _Float16 v16h;
typedef __attribute__((ext_vector_type(8)))  _Float16 v8h;
typedef __attribute__((ext_vector_type(8)))  float    v8f;

#define WMMA_F16(a, b, c) __builtin_amdgcn_wmma_f32_16x16x32_f16(false, (a), false, (b), (short)0, (c), false, false)

// 1/sqrt(64) * log2(e): folded into Q so softmax exponentials are raw exp2.
#define QSCALE 0.18033688011112042f

// Load 16 halves as two contiguous 8-half (16B) chunks -> one v16h fragment.
__device__ __forceinline__ v16h ld16(const _Float16* p0, const _Float16* p1) {
    v8h a = *(const v8h*)p0;
    v8h b = *(const v8h*)p1;
    v16h r;
#pragma unroll
    for (int i = 0; i < 8; ++i) { r[i] = a[i]; r[i + 8] = b[i]; }
    return r;
}

// ---------------------------------------------------------------------------
// Kernel 1: repack Wq/Wk/Wv (768x64 f32 row-major) into f16 WMMA B-fragment
// order: wpack[w][ntile(4)][kchunk(24)][lane(32)][half(16)].
// B-fragment mapping: lane = n + 16g, half h -> K = kchunk*32 + h + 16g.
// ---------------------------------------------------------------------------
__global__ __launch_bounds__(256) void pack_w_kernel(
    const float* __restrict__ Wq, const float* __restrict__ Wk,
    const float* __restrict__ Wv, _Float16* __restrict__ wpack)
{
    int idx = blockIdx.x * 256 + threadIdx.x;
    const int total = 3 * 4 * 24 * 32 * 16;
    if (idx >= total) return;
    int h    = idx & 15;
    int t    = idx >> 4;
    int lane = t & 31;  t >>= 5;
    int kc   = t % 24;  t /= 24;
    int nt   = t & 3;
    int w    = t >> 2;
    int g = lane >> 4, n = lane & 15;
    int k   = kc * 32 + h + 16 * g;   // input-feature index (0..767)
    int col = nt * 16 + n;            // output column (0..63)
    const float* W = (w == 0) ? Wq : ((w == 1) ? Wk : Wv);
    wpack[idx] = (_Float16)W[k * 64 + col];
}

// ---------------------------------------------------------------------------
// Kernel 2: fused QKV projection.  One wave computes one 16x16 output tile of
// one of {Q,K,V} = X[16384x768] * W[768x64] + b, K-loop = 24 WMMAs.
//   w==0 -> Qh[row][h] f16, pre-scaled by log2(e)/sqrt(H)
//   w==1 -> Kh[row][h] f16 row-major
//   w==2 -> Vt[b][h][t] f16 (transposed within batch for P*V B-fragments)
// ---------------------------------------------------------------------------
__global__ __launch_bounds__(128) void proj_kernel(
    const float* __restrict__ x, const _Float16* __restrict__ wpack,
    const float* __restrict__ bq, const float* __restrict__ bk,
    const float* __restrict__ bv,
    _Float16* __restrict__ Qh, _Float16* __restrict__ Kh,
    _Float16* __restrict__ Vt)
{
    const int lane = threadIdx.x & 31;
    const int warp = threadIdx.x >> 5;
    const int wid  = blockIdx.x * 4 + warp;      // 0..12287
    const int w    = wid / 4096;                 // which weight (0..2)
    const int rem  = wid % 4096;
    const int mtile = rem >> 2;                  // 0..1023 (16 rows each)
    const int ntile = rem & 3;                   // 0..3   (16 cols each)
    const int g = lane >> 4;
    const int n = lane & 15;

    const float* xr = x + (size_t)(mtile * 16 + n) * 768;  // A row = lane&15

    v8f acc = {};
    for (int kc = 0; kc < 24; ++kc) {
        const int k0 = kc * 32 + 16 * g;
        v16h a;
#pragma unroll
        for (int i = 0; i < 8; ++i) {
            a[i]     = (_Float16)xr[k0 + i];         // K = 16g + 0..7
            a[i + 8] = (_Float16)xr[k0 + 8 + i];     // K = 16g + 8..15
        }
        const _Float16* bp = wpack + ((((size_t)(w * 4 + ntile) * 24 + kc) * 32 + lane) << 4);
        v16h b = ld16(bp, bp + 8);
        acc = WMMA_F16(a, b, acc);
    }

    const int ncol = ntile * 16 + n;
    if (w == 0) {
        const float bias = bq[ncol];
#pragma unroll
        for (int r = 0; r < 8; ++r) {
            int row = mtile * 16 + 8 * g + r;        // C layout: m = r + 8g
            Qh[(size_t)row * 64 + ncol] = (_Float16)((acc[r] + bias) * QSCALE);
        }
    } else if (w == 1) {
        const float bias = bk[ncol];
#pragma unroll
        for (int r = 0; r < 8; ++r) {
            int row = mtile * 16 + 8 * g + r;
            Kh[(size_t)row * 64 + ncol] = (_Float16)(acc[r] + bias);
        }
    } else {
        const float bias = bv[ncol];
        const int bb = (mtile * 16) >> 11;           // batch (tile never crosses)
        const int t0 = (mtile * 16 + 8 * g) & 2047;  // 8 consecutive t values
        v8h pk;
#pragma unroll
        for (int r = 0; r < 8; ++r) pk[r] = (_Float16)(acc[r] + bias);
        *(v8h*)(Vt + ((size_t)(bb * 64 + ncol) * 2048 + t0)) = pk;
    }
}

// ---------------------------------------------------------------------------
// Kernel 3: fused flash attention.  One wave per (batch, 16-query tile).
// Key blocks of 32: S (16x32) via 4 WMMAs, online softmax in exp2 domain
// (row-max via shfl_xor within lane groups of 16; row-SUM via a 5th WMMA
// against an all-ones B-fragment), P -> LDS -> A-fragment, O += P*V via
// 4 WMMAs against Vt.  Final normalize by row sum, fp32 store.
// ---------------------------------------------------------------------------
__global__ __launch_bounds__(128) void attn_kernel(
    const _Float16* __restrict__ Qh, const _Float16* __restrict__ Kh,
    const _Float16* __restrict__ Vt, float* __restrict__ out)
{
    __shared__ _Float16 plds[4][16 * 40];   // 40-half row stride: 16B aligned, bank-spread
    const int lane = threadIdx.x & 31;
    const int warp = threadIdx.x >> 5;
    const int wid  = blockIdx.x * 4 + warp;      // 0..1023
    const int bb    = wid >> 7;                  // batch
    const int qbase = (wid & 127) * 16;          // query tile start
    const int g = lane >> 4;
    const int n = lane & 15;
    _Float16* pt = plds[warp];

    // Q A-fragments (held for whole loop), H split into two K=32 chunks.
    const _Float16* qrow = Qh + (size_t)(bb * 2048 + qbase + n) * 64;
    const v16h aQ0 = ld16(qrow + 16 * g,      qrow + 16 * g + 8);
    const v16h aQ1 = ld16(qrow + 32 + 16 * g, qrow + 32 + 16 * g + 8);

    // Constant all-ones B-fragment: row-sum tile l = P * 1.
    v16h bOnes;
#pragma unroll
    for (int i = 0; i < 16; ++i) bOnes[i] = (_Float16)1.0f;

    v8f o0 = {}, o1 = {}, o2 = {}, o3 = {}, lacc = {};
    float mrow[8];
#pragma unroll
    for (int r = 0; r < 8; ++r) mrow[r] = -3.0e38f;

    const _Float16* Kb = Kh + (size_t)bb * 2048 * 64;
    const _Float16* Vb = Vt + (size_t)bb * 64 * 2048;

    for (int kb = 0; kb < 2048; kb += 32) {
        // K^T B-fragments: keys (kb..kb+15) and (kb+16..kb+31), H chunks 0/1.
        const _Float16* k0r = Kb + (size_t)(kb + n) * 64;
        const _Float16* k1r = Kb + (size_t)(kb + 16 + n) * 64;
        v16h bK00 = ld16(k0r + 16 * g,      k0r + 16 * g + 8);
        v16h bK01 = ld16(k0r + 32 + 16 * g, k0r + 32 + 16 * g + 8);
        v16h bK10 = ld16(k1r + 16 * g,      k1r + 16 * g + 8);
        v16h bK11 = ld16(k1r + 32 + 16 * g, k1r + 32 + 16 * g + 8);

        // Prefetch next key block's K rows (gfx1250 global_prefetch_b8 path).
        if (kb + 32 < 2048) {
            __builtin_prefetch(k0r + 32 * 64, 0, 2);
            __builtin_prefetch(k1r + 32 * 64, 0, 2);
        }

        v8f s0 = {}, s1 = {};
        s0 = WMMA_F16(aQ0, bK00, s0);
        s0 = WMMA_F16(aQ1, bK01, s0);
        s1 = WMMA_F16(aQ0, bK10, s1);
        s1 = WMMA_F16(aQ1, bK11, s1);

        // Online softmax in exp2 domain.  Lane holds column n for rows
        // m = r + 8g; the 16 lanes of a half-group cooperate on a row.
        float p0[8], p1[8];
#pragma unroll
        for (int r = 0; r < 8; ++r) {
            float mx = fmaxf(s0[r], s1[r]);
            mx = fmaxf(mx, __shfl_xor(mx, 1, 16));
            mx = fmaxf(mx, __shfl_xor(mx, 2, 16));
            mx = fmaxf(mx, __shfl_xor(mx, 4, 16));
            mx = fmaxf(mx, __shfl_xor(mx, 8, 16));
            const float mnew = fmaxf(mrow[r], mx);
            const float cf   = exp2f(mrow[r] - mnew);
            mrow[r] = mnew;
            p0[r] = exp2f(s0[r] - mnew);
            p1[r] = exp2f(s1[r] - mnew);
            o0[r] *= cf; o1[r] *= cf; o2[r] *= cf; o3[r] *= cf;
            lacc[r] *= cf;
        }

        // P (16x32) C-layout -> LDS row-major (stride 40 halves).
#pragma unroll
        for (int r = 0; r < 8; ++r) {
            pt[(r + 8 * g) * 40 + n]      = (_Float16)p0[r];
            pt[(r + 8 * g) * 40 + 16 + n] = (_Float16)p1[r];
        }
        asm volatile("s_wait_dscnt 0" ::: "memory");  // same-wave LDS RAW
        // Reload as A-fragment: row = lane&15, K runs contiguous at 16g.
        const _Float16* pr = pt + n * 40;
        v16h aP = ld16(pr + 16 * g, pr + 16 * g + 8);

        // O += P * V ; l += P * 1 (row sums on the matrix pipe, no shuffles).
        {
            const _Float16* v0 = Vb + (size_t)(0 * 16 + n) * 2048 + kb + 16 * g;
            const _Float16* v1 = Vb + (size_t)(1 * 16 + n) * 2048 + kb + 16 * g;
            const _Float16* v2 = Vb + (size_t)(2 * 16 + n) * 2048 + kb + 16 * g;
            const _Float16* v3 = Vb + (size_t)(3 * 16 + n) * 2048 + kb + 16 * g;
            o0   = WMMA_F16(aP, ld16(v0, v0 + 8), o0);
            o1   = WMMA_F16(aP, ld16(v1, v1 + 8), o1);
            o2   = WMMA_F16(aP, ld16(v2, v2 + 8), o2);
            o3   = WMMA_F16(aP, ld16(v3, v3 + 8), o3);
            lacc = WMMA_F16(aP, bOnes, lacc);
        }
    }

    // Normalize and store fp32 output [B][T][64].
#pragma unroll
    for (int r = 0; r < 8; ++r) {
        const float inv = 1.0f / lacc[r];
        float* op = out + (size_t)(bb * 2048 + qbase + r + 8 * g) * 64;
        op[n]      = o0[r] * inv;
        op[16 + n] = o1[r] * inv;
        op[32 + n] = o2[r] * inv;
        op[48 + n] = o3[r] * inv;
    }
}

// ---------------------------------------------------------------------------
extern "C" void kernel_launch(void* const* d_in, const int* in_sizes, int n_in,
                              void* d_out, int out_size, void* d_ws, size_t ws_size,
                              hipStream_t stream) {
    (void)in_sizes; (void)n_in; (void)out_size; (void)ws_size;
    const float* x  = (const float*)d_in[0];
    const float* Wq = (const float*)d_in[1];
    const float* bq = (const float*)d_in[2];
    const float* Wk = (const float*)d_in[3];
    const float* bk = (const float*)d_in[4];
    const float* Wv = (const float*)d_in[5];
    const float* bv = (const float*)d_in[6];
    float* out = (float*)d_out;

    char* ws = (char*)d_ws;
    _Float16* wpack = (_Float16*)(ws);                          // 288 KB
    _Float16* Qh    = (_Float16*)(ws + (1 << 19));              // 2 MB
    _Float16* Kh    = (_Float16*)(ws + (1 << 19) + (2 << 20));  // 2 MB
    _Float16* Vt    = (_Float16*)(ws + (1 << 19) + (4 << 20));  // 2 MB

    const int pack_total = 3 * 4 * 24 * 32 * 16;                // 147456
    pack_w_kernel<<<(pack_total + 255) / 256, 256, 0, stream>>>(Wq, Wk, Wv, wpack);
    proj_kernel<<<3072, 128, 0, stream>>>(x, wpack, bq, bk, bv, Qh, Kh, Vt);
    attn_kernel<<<256, 128, 0, stream>>>(Qh, Kh, Vt, out);
}
